// CrossWarpingModule_53223234732081
// MI455X (gfx1250) — matrix operand order repaired and checked by reading, not compiled, not run on hardware
//
#include <hip/hip_runtime.h>
#include <hip/hip_bf16.h>
#include <math.h>

// ---------------------------------------------------------------------------
// CrossWarpingModule for MI455X (gfx1250).
// Memory-bound problem (~150MB traffic, ~6-7us at 23.3 TB/s HBM; all
// intermediates fit in 192MB L2). QKV projection (64ch -> 8ch GEMM, K=64)
// runs on v_wmma_f32_16x16x32_f16 (f16 in / f32 accumulate is ample
// precision for 0.05-scale weights).
// ---------------------------------------------------------------------------

#define Bsz   2
#define CH    64
#define FH    256
#define FW    256
#define SB    8          // 4 sub-grids * B
#define H2    128
#define W2    128
#define HID   8
#define PIX_PER_SUB (H2*W2)          // 16384
#define NPIX  (SB*PIX_PER_SUB)       // 131072
#define QKV_ELEMS ((size_t)SB*HID*H2*W2)  // 1,048,576 floats per buffer

typedef __attribute__((ext_vector_type(16))) _Float16 v16h;
typedef __attribute__((ext_vector_type(8)))  float    v8f;

// sub-grid s -> full-image pixel offsets: s0=(::2,::2) s1=(1::2,1::2)
// s2=(::2,1::2) s3=(1::2,::2)
__device__ __forceinline__ int sub_dy(int s) { return (s == 1 || s == 3) ? 1 : 0; }
__device__ __forceinline__ int sub_dx(int s) { return (s == 1 || s == 2) ? 1 : 0; }

// --- WMMA lane layouts (CDNA5 ISA 7.12.2, wave32) ---
// 16-bit A matrix 16x32: lane L holds row M = L&15; element e (packed halves,
// e = 2*vgpr + half) holds K = (L>=16 ? 8:0) + e + (e>=8 ? 8:0).
__device__ __forceinline__ int a_kmap(int lane, int e) {
    return ((lane >= 16) ? 8 : 0) + e + ((e >= 8) ? 8 : 0);
}
// 16-bit B matrix 32x16: lane L holds col N = L&15; element e holds
// K = (L>=16 ? 8:0) + e + (e>=8 ? 8:0)   (same K order as A).
__device__ __forceinline__ int b_kmap(int lane, int e) {
    return ((lane >= 16) ? 8 : 0) + e + ((e >= 8) ? 8 : 0);
}
// f32 C/D 16x16: VGPR r, lane L -> M = r + (L>=16 ? 8:0), N = L&15.

// ---------------------------------------------------------------------------
// Kernel 1: q/k/v projection via WMMA. One wave = 16 pixels x 16 columns.
//   product 1: A = cur tile,  B cols 0..7 = Wq           -> q
//   product 2: A = ref tile,  B cols 0..7 = Wk, 8..15=Wv -> k | v
// K=64 split into two K=32 chunks -> 4 WMMAs per wave.
// Writes q/k/v as [sb,head,h,w] AND transposed [sb,head,w,h] so both
// attention directions get contiguous line loads.
// ---------------------------------------------------------------------------
__global__ void __launch_bounds__(256)
qkv_kernel(const float* __restrict__ cur, const float* __restrict__ ref,
           const float* __restrict__ Wq,  const float* __restrict__ Wk,
           const float* __restrict__ Wv,
           float* __restrict__ qb,  float* __restrict__ kb,  float* __restrict__ vb,
           float* __restrict__ qTb, float* __restrict__ kTb, float* __restrict__ vTb)
{
    const int lane = threadIdx.x & 31;
    const int wave = threadIdx.x >> 5;
    const int tile = blockIdx.x * 8 + wave;   // 16 consecutive pixels per tile
    const int base = tile * 16;               // 16 | W2, never crosses a row

    // This lane's A-matrix row = pixel (base + m)
    const int m   = lane & 15;
    const int pix = base + m;
    const int sb  = pix >> 14;                // /16384
    const int rem = pix & 16383;
    const int h   = rem >> 7;
    const int w   = rem & 127;
    const int s   = sb >> 1;                  // B == 2
    const int b   = sb & 1;
    const int Y   = 2*h + sub_dy(s);
    const int X   = 2*w + sub_dx(s);
    const long long pbase = (long long)b * CH * (FH*FW) + (long long)Y * FW + X;

    v16h a_cur0, a_cur1, a_ref0, a_ref1;
    #pragma unroll
    for (int e = 0; e < 16; ++e) {
        const int k0 = a_kmap(lane, e);       // 0..31
        const int k1 = k0 + 32;               // 32..63
        a_cur0[e] = (_Float16)cur[pbase + (long long)k0 * (FH*FW)];
        a_cur1[e] = (_Float16)cur[pbase + (long long)k1 * (FH*FW)];
        a_ref0[e] = (_Float16)ref[pbase + (long long)k0 * (FH*FW)];
        a_ref1[e] = (_Float16)ref[pbase + (long long)k1 * (FH*FW)];
    }

    // B matrices (2KB weights, L2/K$ resident)
    const int n = lane & 15;                  // output column
    v16h bq0, bq1, bkv0, bkv1;
    #pragma unroll
    for (int e = 0; e < 16; ++e) {
        const int k0 = b_kmap(lane, e);
        const int k1 = k0 + 32;
        bq0[e]  = (n < HID) ? (_Float16)Wq[n*CH + k0] : (_Float16)0.0f;
        bq1[e]  = (n < HID) ? (_Float16)Wq[n*CH + k1] : (_Float16)0.0f;
        bkv0[e] = (n < HID) ? (_Float16)Wk[n*CH + k0] : (_Float16)Wv[(n-HID)*CH + k0];
        bkv1[e] = (n < HID) ? (_Float16)Wk[n*CH + k1] : (_Float16)Wv[(n-HID)*CH + k1];
    }

    v8f accq  = {};
    v8f acckv = {};
    accq  = __builtin_amdgcn_wmma_f32_16x16x32_f16(false, a_cur0, false, bq0,
                                                   (short)0, accq,  false, false);
    accq  = __builtin_amdgcn_wmma_f32_16x16x32_f16(false, a_cur1, false, bq1,
                                                   (short)0, accq,  false, false);
    acckv = __builtin_amdgcn_wmma_f32_16x16x32_f16(false, a_ref0, false, bkv0,
                                                   (short)0, acckv, false, false);
    acckv = __builtin_amdgcn_wmma_f32_16x16x32_f16(false, a_ref1, false, bkv1,
                                                   (short)0, acckv, false, false);

    // Scatter D: VGPR r, this lane -> pixel base + r + (lane>=16)*8, column n.
    const int hi = (lane >= 16) ? 8 : 0;
    #pragma unroll
    for (int r = 0; r < 8; ++r) {
        const int p   = base + r + hi;
        const int sb2 = p >> 14;
        const int rm  = p & 16383;
        const int hh  = rm >> 7;
        const int ww  = rm & 127;
        if (n < HID) {
            const int idx  = ((sb2*HID + n)*H2 + hh)*W2 + ww;   // [sb,head,h,w]
            const int idxT = ((sb2*HID + n)*W2 + ww)*H2 + hh;   // [sb,head,w,h]
            qb [idx]  = accq[r];
            qTb[idxT] = accq[r];
            kb [idx]  = acckv[r];
            kTb[idxT] = acckv[r];
        } else {
            const int head = n - HID;
            const int idx  = ((sb2*HID + head)*H2 + hh)*W2 + ww;
            const int idxT = ((sb2*HID + head)*W2 + ww)*H2 + hh;
            vb [idx]  = acckv[r];
            vTb[idxT] = acckv[r];
        }
    }
}

// ---------------------------------------------------------------------------
// Kernels 2/3: axial attention along one line of 128, all 8 heads (c=1),
// fused with the 8->1 head mix. Buffers are laid out [sb, head, X, L] where
// X indexes the independent line and L the attended dimension:
//   horizontal: normal q/k/v,   X = h, L = w, out[sb][X][t]
//   vertical:   qT/kT/vT,       X = w, L = h, out[sb][t][X]  (TRANS_OUT)
// ---------------------------------------------------------------------------
template <bool TRANS_OUT>
__global__ void __launch_bounds__(128)
attn_kernel(const float* __restrict__ qB, const float* __restrict__ kB,
            const float* __restrict__ vB, const float* __restrict__ Wmix,
            float* __restrict__ obuf)
{
    __shared__ float sk[HID][128];
    __shared__ float sv[HID][128];
    __shared__ float sw[HID];

    const int t  = threadIdx.x;        // query index along the line
    const int sb = blockIdx.x >> 7;
    const int Xi = blockIdx.x & 127;

    #pragma unroll
    for (int head = 0; head < HID; ++head) {
        const int idx = ((sb*HID + head)*128 + Xi)*128 + t;
        sk[head][t] = kB[idx];
        sv[head][t] = vB[idx];
    }
    if (t < HID) sw[t] = Wmix[t];
    __syncthreads();

    float out = 0.0f;
    #pragma unroll
    for (int head = 0; head < HID; ++head) {
        const float qv = qB[((sb*HID + head)*128 + Xi)*128 + t];
        float mx = -1e30f;
        for (int l = 0; l < 128; ++l)
            mx = fmaxf(mx, qv * sk[head][l]);
        float den = 0.0f, num = 0.0f;
        for (int l = 0; l < 128; ++l) {
            const float e = __expf(qv * sk[head][l] - mx);
            den += e;
            num = fmaf(e, sv[head][l], num);
        }
        out = fmaf(sw[head], num / den, out);
    }

    const int oidx = TRANS_OUT ? (sb*PIX_PER_SUB + t*W2 + Xi)
                               : (sb*PIX_PER_SUB + Xi*W2 + t);
    obuf[oidx] = out;
}

// ---------------------------------------------------------------------------
// Kernel 4: bilinear border-clamped warp, indexed by OUTPUT coordinates so
// stores are fully coalesced; gathers hit L2 (ref is 33MB << 192MB).
// Normalization cancels exactly: ix = w + o_hor, iy = h + o_ver.
// ---------------------------------------------------------------------------
__global__ void __launch_bounds__(256)
warp_kernel(const float* __restrict__ ref, const float* __restrict__ oh,
            const float* __restrict__ ov, float* __restrict__ out)
{
    const long long gid = (long long)blockIdx.x * blockDim.x + threadIdx.x;
    const int x = (int)(gid & 255);
    const int y = (int)((gid >> 8) & 255);
    const int c = (int)((gid >> 16) & 63);
    const int b = (int)(gid >> 22);
    if (b >= Bsz) return;

    const int py = y & 1, px = x & 1;
    const int s  = py ? (px ? 1 : 3) : (px ? 2 : 0);
    const int h  = y >> 1, w = x >> 1;
    const int sb = s * Bsz + b;

    const int fidx = sb * PIX_PER_SUB + h * W2 + w;
    const float ix = (float)w + oh[fidx];
    const float iy = (float)h + ov[fidx];

    const float x0f = floorf(ix), y0f = floorf(iy);
    const float wx = ix - x0f,    wy = iy - y0f;
    const int x0 = (int)x0f, y0 = (int)y0f;
    const int x0c = min(max(x0,     0), W2 - 1);
    const int x1c = min(max(x0 + 1, 0), W2 - 1);
    const int y0c = min(max(y0,     0), H2 - 1);
    const int y1c = min(max(y0 + 1, 0), H2 - 1);

    const int dy = sub_dy(s), dx = sub_dx(s);
    const long long cb = (long long)(b * CH + c) * (FH * FW);
    const float v00 = ref[cb + (long long)(2*y0c + dy)*FW + (2*x0c + dx)];
    const float v01 = ref[cb + (long long)(2*y0c + dy)*FW + (2*x1c + dx)];
    const float v10 = ref[cb + (long long)(2*y1c + dy)*FW + (2*x0c + dx)];
    const float v11 = ref[cb + (long long)(2*y1c + dy)*FW + (2*x1c + dx)];

    const float r0 = v00 + wx * (v01 - v00);
    const float r1 = v10 + wx * (v11 - v10);
    out[gid] = r0 + wy * (r1 - r0);
}

// ---------------------------------------------------------------------------
extern "C" void kernel_launch(void* const* d_in, const int* in_sizes, int n_in,
                              void* d_out, int out_size, void* d_ws, size_t ws_size,
                              hipStream_t stream) {
    (void)in_sizes; (void)n_in; (void)out_size; (void)ws_size;
    const float* cur  = (const float*)d_in[0];
    const float* ref  = (const float*)d_in[1];
    const float* Wq   = (const float*)d_in[2];
    const float* Wk   = (const float*)d_in[3];
    const float* Wv   = (const float*)d_in[4];
    const float* Wver = (const float*)d_in[5];
    const float* Whor = (const float*)d_in[6];

    float* ws = (float*)d_ws;          // 25 MB of float scratch
    float* qb  = ws + 0 * QKV_ELEMS;
    float* kb  = ws + 1 * QKV_ELEMS;
    float* vb  = ws + 2 * QKV_ELEMS;
    float* qTb = ws + 3 * QKV_ELEMS;
    float* kTb = ws + 4 * QKV_ELEMS;
    float* vTb = ws + 5 * QKV_ELEMS;
    float* ov  = ws + 6 * QKV_ELEMS;                  // [sb,h,w]
    float* oh  = ov + (size_t)SB * PIX_PER_SUB;       // [sb,h,w]

    // 131072 pixels / 16 per wave / 8 waves per block = 1024 blocks
    qkv_kernel<<<NPIX / 16 / 8, 256, 0, stream>>>(cur, ref, Wq, Wk, Wv,
                                                  qb, kb, vb, qTb, kTb, vTb);
    attn_kernel<true ><<<SB * W2, 128, 0, stream>>>(qTb, kTb, vTb, Wver, ov);
    attn_kernel<false><<<SB * H2, 128, 0, stream>>>(qb,  kb,  vb,  Whor, oh);
    warp_kernel<<<(Bsz * CH * FH * FW) / 256, 256, 0, stream>>>(ref, oh, ov,
                                                                (float*)d_out);
}
